// GGUFLinear_9723805958718
// MI455X (gfx1250) — matrix operand order repaired and checked by reading, not compiled, not run on hardware
//
#include <hip/hip_runtime.h>

// GGUF Q4_0 linear, out[8192,14336] = x[8192,4096] * W^T.  COMPUTE-BOUND
// (AI ~1580 FLOP/B >> balance ~200), so minimize non-WMMA issue:
//  pass 1a: x f32 -> f16 once (64 MiB ws)      [kills 112x re-convert]
//  pass 1b: W Q4_0 -> f16 once (112 MiB ws)    [kills 32x re-dequant]
//  pass 2 : pure f16 GEMM, 256x128 tile, 512 thr = 16 waves (4 waves/SIMD
//           -> full VGPR budget, NO accumulator spills), double-buffered
//           LDS, one barrier per K-step, v_wmma_f32_16x16x32_f16.
// Fallback (ws too small): fused dequant GEMM (v_perm/0x6400/v_pk_fma_f16).

typedef _Float16 v16h __attribute__((ext_vector_type(16)));
typedef _Float16 v8h  __attribute__((ext_vector_type(8)));
typedef _Float16 v4h  __attribute__((ext_vector_type(4)));
typedef _Float16 h2   __attribute__((ext_vector_type(2)));
typedef float    v8f  __attribute__((ext_vector_type(8)));

#define IN_F    4096
#define OUT_F   14336
#define M_ROWS  8192
#define QBYTES  18
#define NBLK    (IN_F / 32)     // 128 Q4_0 blocks per output row
#define KSTEP   32
#define NK      (IN_F / KSTEP)  // 128 K-steps
#define LDA     40              // padded row stride (halves): conflict-free b128
#define MT      256
#define NT      128
#define THREADS 512             // 16 waves: 4(M) x 4(N), wave tile 64x32

// ---------------------------------------------------------------------------
// Pass 1a: x f32 -> f16 (8 elems/thread)
// ---------------------------------------------------------------------------
__global__ __launch_bounds__(256) void x_to_f16(const float* __restrict__ x,
                                                _Float16* __restrict__ xh) {
    const size_t i = ((size_t)blockIdx.x * 256 + threadIdx.x) * 8;
    const float4 a = *(const float4*)(x + i);
    const float4 b = *(const float4*)(x + i + 4);
    const v8h h = { (_Float16)a.x, (_Float16)a.y, (_Float16)a.z, (_Float16)a.w,
                    (_Float16)b.x, (_Float16)b.y, (_Float16)b.z, (_Float16)b.w };
    *(v8h*)(xh + i) = h;
}

// ---------------------------------------------------------------------------
// Pass 1b: W Q4_0 -> f16 row-major [o][k] (one 18B block / thread)
// ---------------------------------------------------------------------------
__global__ __launch_bounds__(256) void w_to_f16(const unsigned char* __restrict__ w,
                                                _Float16* __restrict__ wh) {
    const size_t g = (size_t)blockIdx.x * 256 + threadIdx.x;   // block id
    const size_t o  = g >> 7;          // output row
    const size_t kb = g & 127;         // k-block in row
    const unsigned char* p = w + g * QBYTES;
    _Float16 scale;
    __builtin_memcpy(&scale, p, 2);
    const h2 s2 = { scale, scale };
    const _Float16 nb = scale * (_Float16)(-1032.0f);
    const h2 nb2 = { nb, nb };
    _Float16* dst = wh + o * IN_F + kb * 32;
    #pragma unroll
    for (int j = 0; j < 4; ++j) {
        unsigned int q;
        __builtin_memcpy(&q, p + 2 + 4 * j, 4);
        const unsigned int lo = q & 0x0f0f0f0fu;          // elems 4j..4j+3
        const unsigned int hi = (q >> 4) & 0x0f0f0f0fu;   // elems 16+4j..
        const unsigned int d01 = __builtin_amdgcn_perm(lo, lo, 0x0C010C00u) | 0x64006400u;
        const unsigned int d23 = __builtin_amdgcn_perm(lo, lo, 0x0C030C02u) | 0x64006400u;
        const unsigned int e01 = __builtin_amdgcn_perm(hi, hi, 0x0C010C00u) | 0x64006400u;
        const unsigned int e23 = __builtin_amdgcn_perm(hi, hi, 0x0C030C02u) | 0x64006400u;
        const h2 r01 = __builtin_bit_cast(h2, d01) * s2 + nb2;   // s*(n-8)
        const h2 r23 = __builtin_bit_cast(h2, d23) * s2 + nb2;
        const h2 q01 = __builtin_bit_cast(h2, e01) * s2 + nb2;
        const h2 q23 = __builtin_bit_cast(h2, e23) * s2 + nb2;
        *(v4h*)&dst[4 * j]      = __builtin_shufflevector(r01, r23, 0, 1, 2, 3);
        *(v4h*)&dst[16 + 4 * j] = __builtin_shufflevector(q01, q23, 0, 1, 2, 3);
    }
}

// ---------------------------------------------------------------------------
// Pass 2: pure f16 GEMM, 256x128 tile, 512 threads (spill-free geometry)
// ---------------------------------------------------------------------------
__global__ __launch_bounds__(THREADS) void f16_wmma_gemm(
    const _Float16* __restrict__ xh,
    const _Float16* __restrict__ wh,
    float* __restrict__ out)
{
    __shared__ __align__(16) _Float16 As[2][MT * LDA];   // 40 KiB
    __shared__ __align__(16) _Float16 Bs[2][NT * LDA];   // 20 KiB

    const int tid  = threadIdx.x;
    const int lane = tid & 31;
    const int wave = tid >> 5;          // 0..15
    const int wm   = wave >> 2;         // 0..3 : 64-row stripe
    const int wn   = wave & 3;          // 0..3 : 32-col stripe
    const int l16  = lane & 15;
    const int lh   = lane >> 4;

    const int m_base = blockIdx.y * MT;
    const int n_base = blockIdx.x * NT;

    v8f acc[4][2] = {};

    // A staging: 2 thr/row, 16 halves each;  B staging: 4 thr/row, 8 halves
    const int arow = tid >> 1;
    const int ac   = (tid & 1) << 4;
    const int brow = tid >> 2;
    const int bc   = (tid & 3) << 3;

    auto stage = [&](int buf, int kb) {
        const int k0 = kb * KSTEP;
        const _Float16* ap = xh + (size_t)(m_base + arow) * IN_F + k0 + ac;
        const _Float16* bp = wh + (size_t)(n_base + brow) * IN_F + k0 + bc;
        const v8h a0 = *(const v8h*)ap;
        const v8h a1 = *(const v8h*)(ap + 8);
        const v8h b  = *(const v8h*)bp;
        __builtin_prefetch(ap + KSTEP, 0, 3);
        __builtin_prefetch(bp + KSTEP, 0, 3);
        *(v8h*)&As[buf][arow * LDA + ac]     = a0;   // ds_store_b128
        *(v8h*)&As[buf][arow * LDA + ac + 8] = a1;
        *(v8h*)&Bs[buf][brow * LDA + bc]     = b;
    };

    auto compute = [&](int buf) {
        v16h bf[2];
        #pragma unroll
        for (int ni = 0; ni < 2; ++ni) {
            // B 32x16: lane holds 16 contiguous K for col (lane&15), half by lh
            const int n = wn * 32 + ni * 16 + l16;
            const _Float16* bp = &Bs[buf][n * LDA + lh * 16];
            const v8h b0 = *(const v8h*)bp;
            const v8h b1 = *(const v8h*)(bp + 8);
            bf[ni] = __builtin_shufflevector(b0, b1,
                     0,1,2,3,4,5,6,7,8,9,10,11,12,13,14,15);
        }
        #pragma unroll
        for (int mi = 0; mi < 4; ++mi) {
            // A 16x32: lanes 0-15 K 0-15; lanes 16-31 K 16-31
            const int m = wm * 64 + mi * 16 + l16;
            const _Float16* ap = &As[buf][m * LDA + lh * 16];
            const v8h a0 = *(const v8h*)ap;
            const v8h a1 = *(const v8h*)(ap + 8);
            const v16h af = __builtin_shufflevector(a0, a1,
                            0,1,2,3,4,5,6,7,8,9,10,11,12,13,14,15);
            acc[mi][0] = __builtin_amdgcn_wmma_f32_16x16x32_f16(
                false, af, false, bf[0], (short)0, acc[mi][0], false, false);
            acc[mi][1] = __builtin_amdgcn_wmma_f32_16x16x32_f16(
                false, af, false, bf[1], (short)0, acc[mi][1], false, false);
        }
    };

    stage(0, 0);
    for (int kb = 0; kb < NK; ++kb) {
        __syncthreads();
        if (kb + 1 < NK) stage((kb + 1) & 1, kb + 1);
        compute(kb & 1);
    }

    #pragma unroll
    for (int mi = 0; mi < 4; ++mi) {
        #pragma unroll
        for (int ni = 0; ni < 2; ++ni) {
            const int row0 = m_base + wm * 64 + mi * 16 + lh * 8;
            const int col  = n_base + wn * 32 + ni * 16 + l16;
            float* op = out + (size_t)row0 * OUT_F + col;
            #pragma unroll
            for (int r = 0; r < 8; ++r)
                op[(size_t)r * OUT_F] = acc[mi][ni][r];
        }
    }
}

// ---------------------------------------------------------------------------
// Fallback: fused dequant GEMM (ws too small). 256x128 tile, 512 thr.
// ---------------------------------------------------------------------------
__global__ __launch_bounds__(THREADS) void q40_wmma_gemm(
    const float* __restrict__ x,
    const unsigned char* __restrict__ w,
    float* __restrict__ out)
{
    __shared__ __align__(16) _Float16 As[2][MT * LDA];
    __shared__ __align__(16) _Float16 Bs[2][NT * LDA];

    const int tid  = threadIdx.x;
    const int lane = tid & 31;
    const int wave = tid >> 5;
    const int wm   = wave >> 2;
    const int wn   = wave & 3;
    const int l16  = lane & 15;
    const int lh   = lane >> 4;

    const int m_base = blockIdx.y * MT;
    const int n_base = blockIdx.x * NT;

    v8f acc[4][2] = {};

    auto stage = [&](int buf, int kb) {
        const int k0 = kb * KSTEP;
        #pragma unroll
        for (int it = 0; it < 4; ++it) {
            const int f   = tid + it * THREADS;
            const int row = f >> 3;
            const int c4  = (f & 7) << 2;
            const float4 v = *(const float4*)(x + (size_t)(m_base + row) * IN_F + k0 + c4);
            const v4h hv = { (_Float16)v.x, (_Float16)v.y,
                             (_Float16)v.z, (_Float16)v.w };
            *(v4h*)&As[buf][row * LDA + c4] = hv;
        }
        {
            const int n = tid >> 2;
            const int j = tid & 3;
            const size_t boff = ((size_t)(n_base + n) * NBLK + kb) * QBYTES;
            _Float16 scale;
            __builtin_memcpy(&scale, w + boff, 2);
            unsigned int q;
            __builtin_memcpy(&q, w + boff + 2 + 4 * j, 4);
            __builtin_prefetch(w + boff + QBYTES, 0, 3);
            const h2 s2 = { scale, scale };
            const _Float16 nb = scale * (_Float16)(-1032.0f);
            const h2 nb2 = { nb, nb };
            const unsigned int lo = q & 0x0f0f0f0fu;
            const unsigned int hi = (q >> 4) & 0x0f0f0f0fu;
            const unsigned int d01 = __builtin_amdgcn_perm(lo, lo, 0x0C010C00u) | 0x64006400u;
            const unsigned int d23 = __builtin_amdgcn_perm(lo, lo, 0x0C030C02u) | 0x64006400u;
            const unsigned int e01 = __builtin_amdgcn_perm(hi, hi, 0x0C010C00u) | 0x64006400u;
            const unsigned int e23 = __builtin_amdgcn_perm(hi, hi, 0x0C030C02u) | 0x64006400u;
            const h2 r01 = __builtin_bit_cast(h2, d01) * s2 + nb2;
            const h2 r23 = __builtin_bit_cast(h2, d23) * s2 + nb2;
            const h2 q01 = __builtin_bit_cast(h2, e01) * s2 + nb2;
            const h2 q23 = __builtin_bit_cast(h2, e23) * s2 + nb2;
            _Float16* dst = &Bs[buf][n * LDA];
            *(v4h*)&dst[4 * j]      = __builtin_shufflevector(r01, r23, 0, 1, 2, 3);
            *(v4h*)&dst[16 + 4 * j] = __builtin_shufflevector(q01, q23, 0, 1, 2, 3);
        }
    };

    auto compute = [&](int buf) {
        v16h bf[2];
        #pragma unroll
        for (int ni = 0; ni < 2; ++ni) {
            const int n = wn * 32 + ni * 16 + l16;
            const _Float16* bp = &Bs[buf][n * LDA + lh * 16];
            const v8h b0 = *(const v8h*)bp;
            const v8h b1 = *(const v8h*)(bp + 8);
            bf[ni] = __builtin_shufflevector(b0, b1,
                     0,1,2,3,4,5,6,7,8,9,10,11,12,13,14,15);
        }
        #pragma unroll
        for (int mi = 0; mi < 4; ++mi) {
            const int m = wm * 64 + mi * 16 + l16;
            const _Float16* ap = &As[buf][m * LDA + lh * 16];
            const v8h a0 = *(const v8h*)ap;
            const v8h a1 = *(const v8h*)(ap + 8);
            const v16h af = __builtin_shufflevector(a0, a1,
                            0,1,2,3,4,5,6,7,8,9,10,11,12,13,14,15);
            acc[mi][0] = __builtin_amdgcn_wmma_f32_16x16x32_f16(
                false, af, false, bf[0], (short)0, acc[mi][0], false, false);
            acc[mi][1] = __builtin_amdgcn_wmma_f32_16x16x32_f16(
                false, af, false, bf[1], (short)0, acc[mi][1], false, false);
        }
    };

    stage(0, 0);
    for (int kb = 0; kb < NK; ++kb) {
        __syncthreads();
        if (kb + 1 < NK) stage((kb + 1) & 1, kb + 1);
        compute(kb & 1);
    }

    #pragma unroll
    for (int mi = 0; mi < 4; ++mi) {
        #pragma unroll
        for (int ni = 0; ni < 2; ++ni) {
            const int row0 = m_base + wm * 64 + mi * 16 + lh * 8;
            const int col  = n_base + wn * 32 + ni * 16 + l16;
            float* op = out + (size_t)row0 * OUT_F + col;
            #pragma unroll
            for (int r = 0; r < 8; ++r)
                op[(size_t)r * OUT_F] = acc[mi][ni][r];
        }
    }
}

// ---------------------------------------------------------------------------
extern "C" void kernel_launch(void* const* d_in, const int* in_sizes, int n_in,
                              void* d_out, int out_size, void* d_ws, size_t ws_size,
                              hipStream_t stream) {
    (void)in_sizes; (void)n_in; (void)out_size;
    const float*         x = (const float*)d_in[0];
    const unsigned char* w = (const unsigned char*)d_in[1];
    float*             out = (float*)d_out;

    const size_t xh_bytes = (size_t)M_ROWS * IN_F * 2;          // 64 MiB
    const size_t wh_bytes = (size_t)OUT_F * IN_F * 2;           // 112 MiB

    if (ws_size >= xh_bytes + wh_bytes) {
        _Float16* xh = (_Float16*)d_ws;
        _Float16* wh = (_Float16*)((char*)d_ws + xh_bytes);
        x_to_f16<<<(M_ROWS * (size_t)IN_F) / (256 * 8), 256, 0, stream>>>(x, xh);
        w_to_f16<<<((size_t)OUT_F * NBLK) / 256, 256, 0, stream>>>(w, wh);
        dim3 grid(OUT_F / NT, M_ROWS / MT);       // 112 x 32
        f16_wmma_gemm<<<grid, THREADS, 0, stream>>>(xh, wh, out);
    } else {
        dim3 grid(OUT_F / NT, M_ROWS / MT);       // 112 x 32
        q40_wmma_gemm<<<grid, THREADS, 0, stream>>>(x, w, out);
    }
}
